// SleepStageClassifier_33578054320738
// MI455X (gfx1250) — compile-verified, matrix operand order checked by reading, not run on hardware
//
#include <hip/hip_runtime.h>
#include <hip/hip_bf16.h>
#include <math.h>

// ---------------------------------------------------------------------------
// SleepStageClassifier for MI455X (gfx1250, wave32).
// GEMMs: v_wmma_f32_16x16x32_f16, 64-row register blocking (4 M-tiles/wave)
// so each weight fragment feeds 4 WMMAs. dt_proj folded into x_proj via a
// combined 128x128 weight (exact, linear). Scan: one lane per (seq, channel),
// 16-state in registers, B/C broadcast through LDS.
// ---------------------------------------------------------------------------

typedef __attribute__((ext_vector_type(16))) _Float16 v16h;
typedef __attribute__((ext_vector_type(8)))  float    v8f;
typedef __attribute__((ext_vector_type(4)))  float    f32x4;

#define BN_TOTAL 160          // B*N = 8*20 sequences
#define SEQ_L    1500         // W/2 after maxpool
#define M_ROWS   (BN_TOTAL * SEQ_L)   // 240000, divisible by 64
#define D_MODEL  64
#define D_INNER  128
#define D_STATE  16
#define N_PROJ   160          // 128 (dt via combined weight) + 16 (B) + 16 (C)

__device__ __forceinline__ float silu_f(float v) {
  // v * sigmoid(v) with single v_rcp_f32 instead of full IEEE divide
  return v * __builtin_amdgcn_rcpf(1.f + __expf(-v));
}
__device__ __forceinline__ float softplus_f(float v) {
  return (v > 20.f) ? v : log1pf(__expf(v));
}

// Load a 16x32 (f16) A/B fragment for v_wmma_f32_16x16x32_f16.
// Layout per CDNA5 ISA 7.12.2:
//   lane<16 : row/col = lane,    K = kb+0..7 (v0..v3), kb+16..23 (v4..v7)
//   lane>=16: row/col = lane-16, K = kb+8..15,          kb+24..31
// Two 16B loads per lane -> global_load_b128.
__device__ __forceinline__ v16h load_frag(const _Float16* __restrict__ base,
                                          int ld, int rc0, int k0, int lane) {
  int rc = rc0 + (lane & 15);
  int kb = k0 + ((lane & 16) ? 8 : 0);
  const f32x4* p = reinterpret_cast<const f32x4*>(base + (size_t)rc * ld + kb);
  union { v16h h; f32x4 f[2]; } u;
  u.f[0] = p[0];   // f16 elements kb+0  .. kb+7
  u.f[1] = p[2];   // f16 elements kb+16 .. kb+23
  return u.h;
}

// ---------------------------------------------------------------------------
// Stage 0: conv1(k=3,pad=1) + bias + relu + maxpool(2) + positional encoding
// ---------------------------------------------------------------------------
__global__ void frontend_kernel(const float* __restrict__ x,     // (160,3000)
                                const float* __restrict__ w,     // (64,1,3)
                                const float* __restrict__ cb,    // (64)
                                float* __restrict__ h) {         // (M,64)
  size_t idx = (size_t)blockIdx.x * blockDim.x + threadIdx.x;    // M*64 total
  int d = idx & 63;
  size_t t = idx >> 6;
  int l  = (int)(t % SEQ_L);
  int bn = (int)(t / SEQ_L);
  const float* xr = x + (size_t)bn * 3000;
  float w0 = w[d * 3 + 0], w1 = w[d * 3 + 1], w2 = w[d * 3 + 2], bb = cb[d];
  auto conv = [&](int p) {
    float v = bb + w1 * xr[p];
    if (p > 0)    v += w0 * xr[p - 1];
    if (p < 2999) v += w2 * xr[p + 1];
    return fmaxf(v, 0.f);
  };
  float m = fmaxf(conv(2 * l), conv(2 * l + 1));
  int j = d >> 1;
  float div = __expf(-(float)(2 * j) * (logf(10000.f) / (float)D_MODEL));
  float arg = (float)l * div;
  float pe = (d & 1) ? __cosf(arg) : __sinf(arg);
  h[idx] = m + pe;
}

// ---------------------------------------------------------------------------
// Weight prep: f32 -> f16, plus combined dt weight  W_dt = dtp_w @ xp_w[:4]
// ---------------------------------------------------------------------------
__global__ void cvt_f16_kernel(const float* __restrict__ src,
                               _Float16* __restrict__ dst, int n) {
  int i = blockIdx.x * blockDim.x + threadIdx.x;
  if (i < n) dst[i] = (_Float16)src[i];
}

__global__ void prep_combw_kernel(const float* __restrict__ xpw,   // (36,128)
                                  const float* __restrict__ dtpw,  // (128,4)
                                  _Float16* __restrict__ o) {      // (160,128)
  int idx = blockIdx.x * blockDim.x + threadIdx.x;                 // 160*128
  int k = idx & 127;
  int n = idx >> 7;
  float v;
  if (n < 128) {                     // dt rows: dtp_w @ xp_w[0:4]
    v = 0.f;
    #pragma unroll
    for (int r = 0; r < 4; ++r) v += dtpw[n * 4 + r] * xpw[r * 128 + k];
  } else if (n < 144) {              // B rows
    v = xpw[(4 + (n - 128)) * 128 + k];
  } else {                           // C rows
    v = xpw[(20 + (n - 144)) * 128 + k];
  }
  o[idx] = (_Float16)v;
}

// ---------------------------------------------------------------------------
// LayerNorm over d=64, output f16. One wave per row (2 elements/lane).
// ---------------------------------------------------------------------------
__global__ void ln_kernel(const float* __restrict__ h,
                          const float* __restrict__ g,
                          const float* __restrict__ b,
                          _Float16* __restrict__ out) {
  int row  = blockIdx.x * (blockDim.x >> 5) + (threadIdx.x >> 5);
  int lane = threadIdx.x & 31;
  const float* hr = h + (size_t)row * D_MODEL;
  float v0 = hr[lane], v1 = hr[lane + 32];
  float s = v0 + v1;
  #pragma unroll
  for (int off = 16; off; off >>= 1) s += __shfl_xor(s, off, 32);
  float mean = s * (1.f / 64.f);
  float d0 = v0 - mean, d1 = v1 - mean;
  float vs = d0 * d0 + d1 * d1;
  #pragma unroll
  for (int off = 16; off; off >>= 1) vs += __shfl_xor(vs, off, 32);
  float inv = rsqrtf(vs * (1.f / 64.f) + 1e-5f);
  out[(size_t)row * D_MODEL + lane]      = (_Float16)(d0 * inv * g[lane] + b[lane]);
  out[(size_t)row * D_MODEL + lane + 32] = (_Float16)(d1 * inv * g[lane + 32] + b[lane + 32]);
}

// ---------------------------------------------------------------------------
// GEMM1: xz[M,256] = hn[M,64] @ in_w[256,64]^T + in_b ; split -> x_in, silu(z)
// Block = 64 rows x 256 cols. 128 threads (4 waves); wave w covers N chunk
// [64w,64w+64). 4 M-tiles per wave: each B fragment feeds 4 WMMAs.
// 32 WMMAs per wave (128 per block).
// ---------------------------------------------------------------------------
__global__ void __launch_bounds__(128)
gemm1_kernel(const _Float16* __restrict__ A,      // (M,64)
             const _Float16* __restrict__ W,      // (256,64)
             const float* __restrict__ bias,      // (256)
             float* __restrict__ xin,             // (M,128)
             float* __restrict__ sz) {            // (M,128) = silu(z)
  int rowB = blockIdx.x * 64;
  int wave = threadIdx.x >> 5, lane = threadIdx.x & 31;
  v8f acc[4][4] = {};                  // [nt][mt]
  for (int k0 = 0; k0 < 64; k0 += 32) {
    v16h a[4];
    #pragma unroll
    for (int mt = 0; mt < 4; ++mt)
      a[mt] = load_frag(A, 64, rowB + mt * 16, k0, lane);
    #pragma unroll
    for (int nt = 0; nt < 4; ++nt) {
      v16h bf = load_frag(W, 64, wave * 64 + nt * 16, k0, lane);
      #pragma unroll
      for (int mt = 0; mt < 4; ++mt)
        acc[nt][mt] = __builtin_amdgcn_wmma_f32_16x16x32_f16(
            false, a[mt], false, bf, (short)0, acc[nt][mt], false, false);
    }
  }
  int nl = lane & 15;
  int mo = 8 * (lane >> 4);
  #pragma unroll
  for (int nt = 0; nt < 4; ++nt) {
    int n = wave * 64 + nt * 16 + nl;
    float bv = bias[n];
    #pragma unroll
    for (int mt = 0; mt < 4; ++mt) {
      #pragma unroll
      for (int r = 0; r < 8; ++r) {
        size_t m = (size_t)(rowB + mt * 16 + mo + r);
        float v = acc[nt][mt][r] + bv;
        if (n < 128) xin[m * 128 + n] = v;
        else         sz [m * 128 + (n - 128)] = silu_f(v);
      }
    }
  }
}

// ---------------------------------------------------------------------------
// Depthwise causal conv (k=3, left pad 2) + bias + SiLU ; emit f32 + f16
// ---------------------------------------------------------------------------
__global__ void dwconv_kernel(const float* __restrict__ xin,   // (M,128)
                              const float* __restrict__ w,     // (128,1,3)
                              const float* __restrict__ b,     // (128)
                              float* __restrict__ xc,
                              _Float16* __restrict__ xc16) {
  size_t idx = (size_t)blockIdx.x * blockDim.x + threadIdx.x;  // M*128
  int d = (int)(idx & 127);
  size_t m = idx >> 7;
  int l = (int)(m % SEQ_L);
  float v = w[d * 3 + 2] * xin[m * 128 + d] + b[d];
  if (l >= 1) v += w[d * 3 + 1] * xin[(m - 1) * 128 + d];
  if (l >= 2) v += w[d * 3 + 0] * xin[(m - 2) * 128 + d];
  v = silu_f(v);
  xc[idx] = v;
  xc16[idx] = (_Float16)v;
}

// ---------------------------------------------------------------------------
// GEMM2: dbc[M,160] = xc[M,128] @ combW[160,128]^T   (dt_lin | B | C)
// Block = 64 rows. 320 threads (10 waves), one 16-wide N tile per wave,
// 4 M-tiles per wave (B fragment reused 4x). 16 WMMAs per wave.
// ---------------------------------------------------------------------------
__global__ void __launch_bounds__(320)
gemm2_kernel(const _Float16* __restrict__ A,      // (M,128)
             const _Float16* __restrict__ W,      // (160,128)
             float* __restrict__ dbc) {           // (M,160)
  int rowB = blockIdx.x * 64;
  int wave = threadIdx.x >> 5, lane = threadIdx.x & 31;
  int n0 = wave * 16;
  v8f acc[4] = {};                     // [mt]
  for (int k0 = 0; k0 < 128; k0 += 32) {
    v16h bf = load_frag(W, 128, n0, k0, lane);
    #pragma unroll
    for (int mt = 0; mt < 4; ++mt) {
      v16h a = load_frag(A, 128, rowB + mt * 16, k0, lane);
      acc[mt] = __builtin_amdgcn_wmma_f32_16x16x32_f16(
          false, a, false, bf, (short)0, acc[mt], false, false);
    }
  }
  int n = n0 + (lane & 15);
  int mo = 8 * (lane >> 4);
  #pragma unroll
  for (int mt = 0; mt < 4; ++mt)
    #pragma unroll
    for (int r = 0; r < 8; ++r)
      dbc[(size_t)(rowB + mt * 16 + mo + r) * N_PROJ + n] = acc[mt][r];
}

// ---------------------------------------------------------------------------
// Selective scan. One block per sequence, one lane per channel d.
// state[16] in registers; B/C (32 floats) broadcast via LDS each step.
// ---------------------------------------------------------------------------
__global__ void __launch_bounds__(128)
scan_kernel(const float* __restrict__ dbc,    // (M,160): [dt_lin|B|C]
            const float* __restrict__ xc,     // (M,128)
            const float* __restrict__ sz,     // (M,128) silu(z)
            const float* __restrict__ A_log,  // (128,16)
            const float* __restrict__ Dp,     // (128)
            const float* __restrict__ dtb,    // (128)
            _Float16* __restrict__ y16) {     // (M,128)
  int bn = blockIdx.x;
  int d  = threadIdx.x;
  __shared__ float Bs[D_STATE], Cs[D_STATE];
  float Aa[D_STATE], st[D_STATE];
  #pragma unroll
  for (int s = 0; s < D_STATE; ++s) {
    Aa[s] = -__expf(A_log[d * D_STATE + s]);
    st[s] = 0.f;
  }
  float db = dtb[d], Dd = Dp[d];
  for (int l = 0; l < SEQ_L; ++l) {
    size_t m = (size_t)bn * SEQ_L + l;
    __syncthreads();                      // prior-step LDS reads complete
    if (threadIdx.x < 32) {
      float v = dbc[m * N_PROJ + 128 + threadIdx.x];
      if (threadIdx.x < 16) Bs[threadIdx.x] = v;
      else                  Cs[threadIdx.x - 16] = v;
    }
    float dtlin = dbc[m * N_PROJ + d];
    float xv    = xc[m * 128 + d];
    __syncthreads();
    float dt  = softplus_f(dtlin + db);
    float dtx = dt * xv;
    float y = 0.f;
    #pragma unroll
    for (int s = 0; s < D_STATE; ++s) {
      float dA = __expf(dt * Aa[s]);
      st[s] = st[s] * dA + dtx * Bs[s];
      y += st[s] * Cs[s];
    }
    float yv = (y + Dd * xv) * sz[m * 128 + d];
    y16[m * 128 + d] = (_Float16)yv;
  }
}

// ---------------------------------------------------------------------------
// GEMM3: h[M,64] += y[M,128] @ out_w[64,128]^T + out_b   (residual)
// Block = 64 rows. 128 threads (4 waves), one 16-wide N tile per wave,
// 4 M-tiles per wave. 16 WMMAs per wave.
// ---------------------------------------------------------------------------
__global__ void __launch_bounds__(128)
gemm3_kernel(const _Float16* __restrict__ A,      // (M,128)
             const _Float16* __restrict__ W,      // (64,128)
             const float* __restrict__ bias,      // (64)
             float* __restrict__ h) {             // (M,64) +=
  int rowB = blockIdx.x * 64;
  int wave = threadIdx.x >> 5, lane = threadIdx.x & 31;
  int n0 = wave * 16;
  v8f acc[4] = {};                     // [mt]
  for (int k0 = 0; k0 < 128; k0 += 32) {
    v16h bf = load_frag(W, 128, n0, k0, lane);
    #pragma unroll
    for (int mt = 0; mt < 4; ++mt) {
      v16h a = load_frag(A, 128, rowB + mt * 16, k0, lane);
      acc[mt] = __builtin_amdgcn_wmma_f32_16x16x32_f16(
          false, a, false, bf, (short)0, acc[mt], false, false);
    }
  }
  int n = n0 + (lane & 15);
  int mo = 8 * (lane >> 4);
  float bv = bias[n];
  #pragma unroll
  for (int mt = 0; mt < 4; ++mt)
    #pragma unroll
    for (int r = 0; r < 8; ++r) {
      size_t m = (size_t)(rowB + mt * 16 + mo + r);
      h[m * D_MODEL + n] += acc[mt][r] + bv;
    }
}

// ---------------------------------------------------------------------------
// Mean pool over L + 5-way classifier head
// ---------------------------------------------------------------------------
__global__ void head_kernel(const float* __restrict__ h,      // (M,64)
                            const float* __restrict__ cw,     // (5,64)
                            const float* __restrict__ cb,     // (5)
                            float* __restrict__ out) {        // (160,5)
  int bn = blockIdx.x;
  __shared__ float pooled[D_MODEL];
  int tid = threadIdx.x;                  // 64 threads
  const float* hp = h + (size_t)bn * SEQ_L * D_MODEL;
  float s = 0.f;
  for (int l = 0; l < SEQ_L; ++l) s += hp[(size_t)l * D_MODEL + tid];
  pooled[tid] = s * (1.f / (float)SEQ_L);
  __syncthreads();
  if (tid < 5) {
    float acc = cb[tid];
    #pragma unroll
    for (int dd = 0; dd < D_MODEL; ++dd) acc += pooled[dd] * cw[tid * D_MODEL + dd];
    out[bn * 5 + tid] = acc;
  }
}

// ---------------------------------------------------------------------------
// Host launcher
// ---------------------------------------------------------------------------
extern "C" void kernel_launch(void* const* d_in, const int* in_sizes, int n_in,
                              void* d_out, int out_size, void* d_ws, size_t ws_size,
                              hipStream_t stream) {
  (void)in_sizes; (void)n_in; (void)out_size; (void)ws_size;
  const float* x       = (const float*)d_in[0];
  const float* conv1_w = (const float*)d_in[1];
  const float* conv1_b = (const float*)d_in[2];
  const float* ln_g    = (const float*)d_in[3];
  const float* ln_b    = (const float*)d_in[4];
  const float* in_w    = (const float*)d_in[5];
  const float* in_b    = (const float*)d_in[6];
  const float* cdw     = (const float*)d_in[7];
  const float* cdb     = (const float*)d_in[8];
  const float* xp_w    = (const float*)d_in[9];
  const float* dtp_w   = (const float*)d_in[10];
  const float* dtp_b   = (const float*)d_in[11];
  const float* A_log   = (const float*)d_in[12];
  const float* Dp      = (const float*)d_in[13];
  const float* out_w   = (const float*)d_in[14];
  const float* out_b   = (const float*)d_in[15];
  const float* cls_w   = (const float*)d_in[16];
  const float* cls_b   = (const float*)d_in[17];
  float* out = (float*)d_out;

  // Workspace carve-up (256B aligned). ~707 MB total.
  char* ws = (char*)d_ws;
  size_t off = 0;
  auto carve = [&](size_t bytes) -> void* {
    off = (off + 255) & ~(size_t)255;
    void* p = ws + off;
    off += bytes;
    return p;
  };
  const size_t M = M_ROWS;
  float*     b_h     = (float*)    carve(M * D_MODEL * sizeof(float));
  _Float16*  b_half  = (_Float16*) carve(M * D_INNER * sizeof(_Float16)); // hn16 then y16
  float*     b_xin   = (float*)    carve(M * D_INNER * sizeof(float));
  float*     b_sz    = (float*)    carve(M * D_INNER * sizeof(float));
  float*     b_xc    = (float*)    carve(M * D_INNER * sizeof(float));
  _Float16*  b_xc16  = (_Float16*) carve(M * D_INNER * sizeof(_Float16));
  float*     b_dbc   = (float*)    carve(M * N_PROJ  * sizeof(float));
  _Float16*  w_in16  = (_Float16*) carve(256 * 64  * sizeof(_Float16));
  _Float16*  w_cmb16 = (_Float16*) carve(N_PROJ * D_INNER * sizeof(_Float16));
  _Float16*  w_out16 = (_Float16*) carve(D_MODEL * D_INNER * sizeof(_Float16));

  // Stage 0: frontend
  frontend_kernel<<<(M * D_MODEL) / 256, 256, 0, stream>>>(x, conv1_w, conv1_b, b_h);

  for (int blk = 0; blk < 2; ++blk) {
    const float* g_b   = ln_g  + blk * D_MODEL;
    const float* bb_b  = ln_b  + blk * D_MODEL;
    const float* iw_b  = in_w  + (size_t)blk * 256 * 64;
    const float* ib_b  = in_b  + blk * 256;
    const float* cw_b  = cdw   + (size_t)blk * D_INNER * 3;
    const float* cb_b  = cdb   + blk * D_INNER;
    const float* xw_b  = xp_w  + (size_t)blk * 36 * D_INNER;
    const float* dw_b  = dtp_w + (size_t)blk * D_INNER * 4;
    const float* db_b  = dtp_b + blk * D_INNER;
    const float* Al_b  = A_log + (size_t)blk * D_INNER * D_STATE;
    const float* Dp_b  = Dp    + blk * D_INNER;
    const float* ow_b  = out_w + (size_t)blk * D_MODEL * D_INNER;
    const float* ob_b  = out_b + blk * D_MODEL;

    // Weight prep (f16 conversions + dt-composition)
    cvt_f16_kernel<<<(256 * 64) / 256, 256, 0, stream>>>(iw_b, w_in16, 256 * 64);
    prep_combw_kernel<<<(N_PROJ * D_INNER) / 256, 256, 0, stream>>>(xw_b, dw_b, w_cmb16);
    cvt_f16_kernel<<<(D_MODEL * D_INNER) / 256, 256, 0, stream>>>(ow_b, w_out16,
                                                                  D_MODEL * D_INNER);
    // Block pipeline
    ln_kernel<<<M / 8, 256, 0, stream>>>(b_h, g_b, bb_b, b_half);
    gemm1_kernel<<<M / 64, 128, 0, stream>>>(b_half, w_in16, ib_b, b_xin, b_sz);
    dwconv_kernel<<<(M * D_INNER) / 256, 256, 0, stream>>>(b_xin, cw_b, cb_b, b_xc, b_xc16);
    gemm2_kernel<<<M / 64, 320, 0, stream>>>(b_xc16, w_cmb16, b_dbc);
    scan_kernel<<<BN_TOTAL, 128, 0, stream>>>(b_dbc, b_xc, b_sz, Al_b, Dp_b, db_b, b_half);
    gemm3_kernel<<<M / 64, 128, 0, stream>>>(b_half, w_out16, ob_b, b_h);
  }

  head_kernel<<<BN_TOTAL, 64, 0, stream>>>(b_h, cls_w, cls_b, out);
}
